// Decoder_73426760892701
// MI455X (gfx1250) — compile-verified, matrix operand order checked by reading, not run on hardware
//
#include <hip/hip_runtime.h>

// ---------------------------------------------------------------------------
// CDNA5 (gfx1250) attention-LSTM decoder.   B=1024, T=128, M=256, P=256.
//   WU_d . z = WU_hd.[h|c]  (per-step tiny GEMM)  +  WU_x.X  (precomputed once)
// GEMMs: v_wmma_f32_16x16x32_bf16, one 16x16 tile per wave32.
// X and Xw kept as bf16 so the 127-step attention stream (128 MB) stays
// resident in the MI455X's 192 MB L2.
// ---------------------------------------------------------------------------

typedef __attribute__((ext_vector_type(16))) __bf16 v16bf;
typedef __attribute__((ext_vector_type(8)))  float  v8f;

union BfFrag { unsigned u[8]; v16bf v; };

#define Bsz   1024
#define Tsz   128
#define Msz   256
#define Psz   256

__device__ __forceinline__ unsigned pack_bf16(float lo, float hi) {
    unsigned a = (__float_as_uint(lo) + 0x8000u) >> 16;
    unsigned b = (__float_as_uint(hi) + 0x8000u) & 0xFFFF0000u;
    return (a & 0xFFFFu) | b;
}
__device__ __forceinline__ unsigned short f32_to_bf16(float x) {
    return (unsigned short)((__float_as_uint(x) + 0x8000u) >> 16);
}
__device__ __forceinline__ float bf16_to_f32(unsigned short x) {
    return __uint_as_float(((unsigned)x) << 16);
}
__device__ __forceinline__ float bf_lo(unsigned u) { return __uint_as_float(u << 16); }
__device__ __forceinline__ float bf_hi(unsigned u) { return __uint_as_float(u & 0xFFFF0000u); }
__device__ __forceinline__ float sigm(float x) { return 1.f / (1.f + __expf(-x)); }

// A-fragment (16x32 bf16) packed on the fly from f32 row-major; rowp = &A[row][k0]
__device__ __forceinline__ void load_frag_f32(BfFrag& f, const float* rowp) {
    int lane = threadIdx.x & 31;
    int kh = (lane >> 4) << 3;                 // 0 or 8
    float4 a0 = *(const float4*)(rowp + kh);
    float4 a1 = *(const float4*)(rowp + kh + 4);
    float4 b0 = *(const float4*)(rowp + 16 + kh);
    float4 b1 = *(const float4*)(rowp + 16 + kh + 4);
    f.u[0] = pack_bf16(a0.x, a0.y); f.u[1] = pack_bf16(a0.z, a0.w);
    f.u[2] = pack_bf16(a1.x, a1.y); f.u[3] = pack_bf16(a1.z, a1.w);
    f.u[4] = pack_bf16(b0.x, b0.y); f.u[5] = pack_bf16(b0.z, b0.w);
    f.u[6] = pack_bf16(b1.x, b1.y); f.u[7] = pack_bf16(b1.z, b1.w);
}
// Fragment (16x32 bf16) from bf16 row-major; rowp = &A[row][k0]
__device__ __forceinline__ void load_frag_bf16(BfFrag& f, const unsigned short* rowp) {
    int lane = threadIdx.x & 31;
    int kh = (lane >> 4) << 3;
    uint4 q0 = *(const uint4*)(rowp + kh);
    uint4 q1 = *(const uint4*)(rowp + 16 + kh);
    f.u[0] = q0.x; f.u[1] = q0.y; f.u[2] = q0.z; f.u[3] = q0.w;
    f.u[4] = q1.x; f.u[5] = q1.y; f.u[6] = q1.z; f.u[7] = q1.w;
}

// --------------------------------------------------------------------------
// f32 -> bf16 strided convert (weights)
// --------------------------------------------------------------------------
__global__ void k_conv(const float* __restrict__ src, unsigned short* __restrict__ dst,
                       int rows, int cols, int src_ld, int src_off) {
    int i = blockIdx.x * blockDim.x + threadIdx.x;
    if (i >= rows * cols) return;
    int r = i / cols, c = i - r * cols;
    dst[i] = f32_to_bf16(src[(size_t)r * src_ld + src_off + c]);
}

// --------------------------------------------------------------------------
// Bulk f32 -> bf16 convert, 8 elements/thread (for X: 33.5M elements)
// --------------------------------------------------------------------------
__global__ void k_convX(const float* __restrict__ src, unsigned short* __restrict__ dst) {
    size_t i = ((size_t)blockIdx.x * blockDim.x + threadIdx.x) * 8;
    float4 a = *(const float4*)(src + i);
    float4 b = *(const float4*)(src + i + 4);
    uint4 q;
    q.x = pack_bf16(a.x, a.y); q.y = pack_bf16(a.z, a.w);
    q.z = pack_bf16(b.x, b.y); q.w = pack_bf16(b.z, b.w);
    *(uint4*)(dst + i) = q;
}

// --------------------------------------------------------------------------
// Xw = X @ WUx^T :  (131072 x 256) x (256 x 256) -> bf16.  One 16x16 tile/wave.
// XBF=true: A loaded from pre-converted bf16 (pure load+wmma inner loop).
// --------------------------------------------------------------------------
template <bool XBF>
__global__ void k_xw(const float* __restrict__ X, const unsigned short* __restrict__ Xbf,
                     const unsigned short* __restrict__ WUx, unsigned short* __restrict__ Xw) {
    int wave = (blockIdx.x * blockDim.x + threadIdx.x) >> 5;
    int tileN = wave & 15;                     // 16 col tiles
    int tileM = wave >> 4;                     // 8192 row tiles
    int lane  = threadIdx.x & 31;
    int r = tileM * 16 + (lane & 15);
    int n = tileN * 16 + (lane & 15);
    v8f acc = {};
    for (int k0 = 0; k0 < Msz; k0 += 32) {
        BfFrag a, b;
        if (XBF) load_frag_bf16(a, Xbf + (size_t)r * Msz + k0);
        else     load_frag_f32 (a, X   + (size_t)r * Msz + k0);
        load_frag_bf16(b, WUx + (size_t)n * Msz + k0);
        acc = __builtin_amdgcn_wmma_f32_16x16x32_bf16(false, a.v, false, b.v,
                                                      (short)0, acc, false, false);
    }
    int col   = tileN * 16 + (lane & 15);
    int rbase = tileM * 16 + ((lane >> 4) << 3);
#pragma unroll
    for (int rr = 0; rr < 8; ++rr)
        Xw[(size_t)(rbase + rr) * Msz + col] = f32_to_bf16(acc[rr]);
}

// --------------------------------------------------------------------------
// A = [h|c] @ WU_hd^T : (1024 x 512) x (512 x 256) -> f32
// --------------------------------------------------------------------------
__global__ void k_attA(const unsigned short* __restrict__ dbf,
                       const unsigned short* __restrict__ WUhd,
                       float* __restrict__ Amat) {
    int wave = (blockIdx.x * blockDim.x + threadIdx.x) >> 5;
    int tileN = wave & 15;                     // 16 col tiles
    int tileM = wave >> 4;                     // 64 row tiles
    int lane  = threadIdx.x & 31;
    int r = tileM * 16 + (lane & 15);
    int n = tileN * 16 + (lane & 15);
    v8f acc = {};
    for (int k0 = 0; k0 < 2 * Psz; k0 += 32) {
        BfFrag a, b;
        load_frag_bf16(a, dbf  + (size_t)r * (2 * Psz) + k0);
        load_frag_bf16(b, WUhd + (size_t)n * (2 * Psz) + k0);
        acc = __builtin_amdgcn_wmma_f32_16x16x32_bf16(false, a.v, false, b.v,
                                                      (short)0, acc, false, false);
    }
    int col   = tileN * 16 + (lane & 15);
    int rbase = tileM * 16 + ((lane >> 4) << 3);
#pragma unroll
    for (int rr = 0; rr < 8; ++rr)
        Amat[(size_t)(rbase + rr) * Msz + col] = acc[rr];
}

// --------------------------------------------------------------------------
// gates = h @ W_hh^T + y_tilde*W_ih^T + b_ih + b_hh : (1024 x 1024) f32
// --------------------------------------------------------------------------
__global__ void k_gates(const unsigned short* __restrict__ dbf,  // [h|c], ld=512; h = first 256
                        const unsigned short* __restrict__ Whh,  // (1024 x 256) bf16
                        const float* __restrict__ y_tilde,
                        const float* __restrict__ W_ih,
                        const float* __restrict__ b_ih,
                        const float* __restrict__ b_hh,
                        float* __restrict__ gates) {
    int wave = (blockIdx.x * blockDim.x + threadIdx.x) >> 5;
    int tileN = wave & 63;                     // 64 col tiles (4P=1024)
    int tileM = wave >> 6;                     // 64 row tiles
    int lane  = threadIdx.x & 31;
    int r = tileM * 16 + (lane & 15);
    int n = tileN * 16 + (lane & 15);
    v8f acc = {};
    for (int k0 = 0; k0 < Psz; k0 += 32) {
        BfFrag a, b;
        load_frag_bf16(a, dbf + (size_t)r * (2 * Psz) + k0);   // h half
        load_frag_bf16(b, Whh + (size_t)n * Psz + k0);
        acc = __builtin_amdgcn_wmma_f32_16x16x32_bf16(false, a.v, false, b.v,
                                                      (short)0, acc, false, false);
    }
    int col   = tileN * 16 + (lane & 15);
    int rbase = tileM * 16 + ((lane >> 4) << 3);
    float wih  = W_ih[col];
    float bias = b_ih[col] + b_hh[col];
#pragma unroll
    for (int rr = 0; rr < 8; ++rr) {
        int row = rbase + rr;
        gates[(size_t)row * (4 * Psz) + col] = acc[rr] + y_tilde[row] * wih + bias;
    }
}

// --------------------------------------------------------------------------
// Fused attention: l -> softmax -> ctx -> y_tilde.  One block (256 thr) per b.
// Lane owns 8 contiguous m (vectorized b128 reads of Xw).
// --------------------------------------------------------------------------
__global__ void k_attn(const float* __restrict__ Amat, const unsigned short* __restrict__ Xw,
                       const unsigned short* __restrict__ Xbf, const float* __restrict__ X,
                       const float* __restrict__ v_d, const float* __restrict__ Wl,
                       const float* __restrict__ Y, float* __restrict__ ctx,
                       float* __restrict__ y_tilde, int step, int use_bf) {
    int b   = blockIdx.x;
    int tid = threadIdx.x;                     // 0..255
    int wv  = tid >> 5, lane = tid & 31;
    __shared__ float l_s[Tsz];
    __shared__ float red_s[256];

    // --- phase 1: l[t] = sum_m v_d[m]*tanh(A[b,m] + Xw[t,b,m]);  lane owns m0..m0+7
    int m0 = lane * 8;
    float vd_m[8], A_m[8];
    {
        float4 v0 = *(const float4*)(v_d + m0);
        float4 v1 = *(const float4*)(v_d + m0 + 4);
        vd_m[0] = v0.x; vd_m[1] = v0.y; vd_m[2] = v0.z; vd_m[3] = v0.w;
        vd_m[4] = v1.x; vd_m[5] = v1.y; vd_m[6] = v1.z; vd_m[7] = v1.w;
        float4 a0 = *(const float4*)(Amat + (size_t)b * Msz + m0);
        float4 a1 = *(const float4*)(Amat + (size_t)b * Msz + m0 + 4);
        A_m[0] = a0.x; A_m[1] = a0.y; A_m[2] = a0.z; A_m[3] = a0.w;
        A_m[4] = a1.x; A_m[5] = a1.y; A_m[6] = a1.z; A_m[7] = a1.w;
    }
    for (int t = wv; t < Tsz; t += 8) {
        uint4 q = *(const uint4*)(Xw + ((size_t)t * Bsz + b) * Msz + m0);
        float xv[8] = { bf_lo(q.x), bf_hi(q.x), bf_lo(q.y), bf_hi(q.y),
                        bf_lo(q.z), bf_hi(q.z), bf_lo(q.w), bf_hi(q.w) };
        float s = 0.f;
#pragma unroll
        for (int i = 0; i < 8; ++i) s += vd_m[i] * tanhf(A_m[i] + xv[i]);
#pragma unroll
        for (int off = 16; off > 0; off >>= 1) s += __shfl_xor(s, off, 32);
        if (lane == 0) l_s[t] = s;
    }
    __syncthreads();

    // --- phase 2: softmax over t (128 values)
    red_s[tid] = (tid < Tsz) ? l_s[tid] : -3.0e38f;
    __syncthreads();
    for (int off = 128; off > 0; off >>= 1) {
        if (tid < off) red_s[tid] = fmaxf(red_s[tid], red_s[tid + off]);
        __syncthreads();
    }
    float mx = red_s[0];
    __syncthreads();
    float e = (tid < Tsz) ? __expf(l_s[tid] - mx) : 0.f;
    red_s[tid] = e;
    __syncthreads();
    for (int off = 128; off > 0; off >>= 1) {
        if (tid < off) red_s[tid] += red_s[tid + off];
        __syncthreads();
    }
    float inv = 1.f / red_s[0];
    __syncthreads();
    if (tid < Tsz) l_s[tid] = e * inv;         // beta
    __syncthreads();

    // --- phase 3: ctx[m] = sum_t beta[t] * X[t,b,m]   (m = tid)
    float acc = 0.f;
    if (use_bf) {
        const unsigned short* xb = Xbf + (size_t)b * Msz + tid;
        for (int t = 0; t < Tsz; ++t) {
            if (t + 4 < Tsz) __builtin_prefetch(xb + (size_t)(t + 4) * Bsz * Msz, 0, 1);
            acc += l_s[t] * bf16_to_f32(xb[(size_t)t * Bsz * Msz]);
        }
    } else {
        const float* xb = X + (size_t)b * Msz + tid;
        for (int t = 0; t < Tsz; ++t) {
            if (t + 4 < Tsz) __builtin_prefetch(xb + (size_t)(t + 4) * Bsz * Msz, 0, 1);
            acc += l_s[t] * xb[(size_t)t * Bsz * Msz];
        }
    }
    ctx[(size_t)b * Msz + tid] = acc;

    // --- phase 4: y_tilde = Wl[0]*y_t + sum_m Wl[1+m]*ctx[m]
    red_s[tid] = Wl[1 + tid] * acc;
    __syncthreads();
    for (int off = 128; off > 0; off >>= 1) {
        if (tid < off) red_s[tid] += red_s[tid + off];
        __syncthreads();
    }
    if (tid == 0) y_tilde[b] = red_s[0] + Wl[0] * Y[(size_t)b * (Tsz - 1) + step];
}

// --------------------------------------------------------------------------
// LSTM elementwise update; refresh bf16 [h|c] matrix for next step's GEMMs.
// --------------------------------------------------------------------------
__global__ void k_lstm(const float* __restrict__ gates, float* __restrict__ h,
                       float* __restrict__ c, unsigned short* __restrict__ dbf) {
    int idx = blockIdx.x * blockDim.x + threadIdx.x;   // 0..262143
    int b = idx >> 8, p = idx & 255;
    const float* g = gates + (size_t)b * (4 * Psz);
    float ig = sigm(g[p]);
    float fg = sigm(g[Psz + p]);
    float gg = tanhf(g[2 * Psz + p]);
    float og = sigm(g[3 * Psz + p]);
    float cn = fg * c[idx] + ig * gg;
    float hn = og * tanhf(cn);
    c[idx] = cn;
    h[idx] = hn;
    dbf[(size_t)b * (2 * Psz) + p]       = f32_to_bf16(hn);
    dbf[(size_t)b * (2 * Psz) + Psz + p] = f32_to_bf16(cn);
}

__global__ void k_init(float* __restrict__ h, float* __restrict__ c,
                       unsigned short* __restrict__ dbf) {
    int i = blockIdx.x * blockDim.x + threadIdx.x;     // 0..524287
    if (i < Bsz * Psz) { h[i] = 0.f; c[i] = 0.f; }
    dbf[i] = 0;
}

// --------------------------------------------------------------------------
// out[b] = vb_w . (Wb_w @ [h|ctx] + Wb_b) + vb_b.  One block per b.
// --------------------------------------------------------------------------
__global__ void k_head(const float* __restrict__ h, const float* __restrict__ ctx,
                       const float* __restrict__ Wb_w, const float* __restrict__ Wb_b,
                       const float* __restrict__ vb_w, const float* __restrict__ vb_b,
                       float* __restrict__ out) {
    int b = blockIdx.x, p = threadIdx.x;               // 256 threads
    __shared__ float red_s[256];
    const float* hrow = h   + (size_t)b * Psz;
    const float* crow = ctx + (size_t)b * Msz;
    const float* w    = Wb_w + (size_t)p * (Psz + Msz);
    float s = Wb_b[p];
    for (int k = 0; k < Psz; ++k) s += w[k] * hrow[k];
    for (int k = 0; k < Msz; ++k) s += w[Psz + k] * crow[k];
    red_s[p] = vb_w[p] * s;
    __syncthreads();
    for (int off = 128; off > 0; off >>= 1) {
        if (p < off) red_s[p] += red_s[p + off];
        __syncthreads();
    }
    if (p == 0) out[b] = red_s[0] + vb_b[0];
}

// --------------------------------------------------------------------------
extern "C" void kernel_launch(void* const* d_in, const int* in_sizes, int n_in,
                              void* d_out, int out_size, void* d_ws, size_t ws_size,
                              hipStream_t stream) {
    const float* Y    = (const float*)d_in[0];    // (1024,127,1)
    const float* X    = (const float*)d_in[1];    // (128,1024,256)
    const float* WU_d = (const float*)d_in[2];    // (256,768)
    const float* v_d  = (const float*)d_in[3];    // (1,256)
    const float* Wl   = (const float*)d_in[4];    // (1,257)
    const float* W_ih = (const float*)d_in[5];    // (1024,1)
    const float* W_hh = (const float*)d_in[6];    // (1024,256)
    const float* b_ih = (const float*)d_in[7];
    const float* b_hh = (const float*)d_in[8];
    const float* Wb_w = (const float*)d_in[9];    // (256,512)
    const float* Wb_b = (const float*)d_in[10];
    const float* vb_w = (const float*)d_in[11];   // (1,256)
    const float* vb_b = (const float*)d_in[12];
    float* out = (float*)d_out;

    // ---- workspace carve ----
    char* ws = (char*)d_ws;
    auto carve = [&](size_t bytes) {
        void* p = (void*)ws;
        ws += (bytes + 255) & ~(size_t)255;
        return p;
    };
    unsigned short* Xw    = (unsigned short*)carve((size_t)Tsz * Bsz * Msz * 2); // 64 MB
    unsigned short* WUx   = (unsigned short*)carve((size_t)Msz * Msz * 2);
    unsigned short* WUhd  = (unsigned short*)carve((size_t)Msz * 2 * Psz * 2);
    unsigned short* Whh   = (unsigned short*)carve((size_t)4 * Psz * Psz * 2);
    unsigned short* dbf   = (unsigned short*)carve((size_t)Bsz * 2 * Psz * 2);
    float* h       = (float*)carve((size_t)Bsz * Psz * 4);
    float* c       = (float*)carve((size_t)Bsz * Psz * 4);
    float* Amat    = (float*)carve((size_t)Bsz * Msz * 4);
    float* ctx     = (float*)carve((size_t)Bsz * Msz * 4);
    float* gates   = (float*)carve((size_t)Bsz * 4 * Psz * 4);
    float* y_tilde = (float*)carve((size_t)Bsz * 4);
    // optional bf16 copy of X (64 MB) — only if workspace allows
    size_t used = (size_t)(ws - (char*)d_ws);
    size_t xbf_bytes = (size_t)Tsz * Bsz * Msz * 2;
    int use_bf = (used + xbf_bytes <= ws_size) ? 1 : 0;
    unsigned short* Xbf = use_bf ? (unsigned short*)carve(xbf_bytes) : (unsigned short*)0;

    // ---- weight conversion to bf16 ----
    k_conv<<<(Msz * Msz + 255) / 256, 256, 0, stream>>>(WU_d, WUx, Msz, Msz, 768, 512);   // WU_x
    k_conv<<<(Msz * 512 + 255) / 256, 256, 0, stream>>>(WU_d, WUhd, Msz, 512, 768, 0);    // WU_hd
    k_conv<<<(1024 * Msz + 255) / 256, 256, 0, stream>>>(W_hh, Whh, 1024, Msz, Msz, 0);   // W_hh

    // ---- X -> bf16 (once), then Xw = X @ WU_x^T (once) ----
    int waves_xw = (Tsz * Bsz / 16) * (Msz / 16);      // 131072
    if (use_bf) {
        size_t nthreads = (size_t)Tsz * Bsz * Msz / 8; // 8 elems/thread
        k_convX<<<(unsigned)(nthreads / 256), 256, 0, stream>>>(X, Xbf);
        k_xw<true><<<waves_xw / 8, 256, 0, stream>>>(X, Xbf, WUx, Xw);
    } else {
        k_xw<false><<<waves_xw / 8, 256, 0, stream>>>(X, Xbf, WUx, Xw);
    }

    // ---- init recurrent state ----
    k_init<<<(Bsz * 2 * Psz) / 256, 256, 0, stream>>>(h, c, dbf);

    // ---- recurrence: 127 steps ----
    for (int s = 0; s < Tsz - 1; ++s) {
        {   // A = [h|c] @ WU_hd^T
            int waves = (Bsz / 16) * (Msz / 16);       // 1024
            k_attA<<<waves / 8, 256, 0, stream>>>(dbf, WUhd, Amat);
        }
        k_attn<<<Bsz, 256, 0, stream>>>(Amat, Xw, Xbf, X, v_d, Wl, Y, ctx, y_tilde, s, use_bf);
        {   // gates = h @ W_hh^T + y_tilde*W_ih + biases
            int waves = (Bsz / 16) * (4 * Psz / 16);   // 4096
            k_gates<<<waves / 8, 256, 0, stream>>>(dbf, Whh, y_tilde, W_ih, b_ih, b_hh, gates);
        }
        k_lstm<<<(Bsz * Psz) / 256, 256, 0, stream>>>(gates, h, c, dbf);
    }

    // ---- output head ----
    k_head<<<Bsz, 256, 0, stream>>>(h, ctx, Wb_w, Wb_b, vb_w, vb_b, out);
}
